// MultiHeadSelfAttention_20186346291428
// MI455X (gfx1250) — matrix operand order repaired and checked by reading, not compiled, not run on hardware
//
#include <hip/hip_runtime.h>

// Problem constants (match reference)
#define B_      4
#define S_      1024
#define D_      512
#define H_      8
#define DK_     64
#define MAXREL_ 10
#define VOCAB_  21

// GEMM tiling
#define MT_ 128
#define NT_ 64
#define KT_ 32

typedef __attribute__((ext_vector_type(16))) __bf16 v16bf;
typedef __attribute__((ext_vector_type(8)))  __bf16 v8bf;
typedef __attribute__((ext_vector_type(8)))  float  v8f;

// Vector-of-4-int type matching the async-to-LDS builtin's parameter type
typedef int vi4_ __attribute__((__vector_size__(4 * sizeof(int))));

// ---------------------------------------------------------------------------
// CDNA5 async global->LDS copy (ASYNCcnt-tracked) with safe fallback
// ---------------------------------------------------------------------------
#if defined(__has_builtin)
#  if __has_builtin(__builtin_amdgcn_global_load_async_to_lds_b128)
#    define HAVE_ASYNC_LDS 1
#  endif
#endif
#ifndef HAVE_ASYNC_LDS
#  define HAVE_ASYNC_LDS 0
#endif

#if HAVE_ASYNC_LDS
#  if __has_builtin(__builtin_amdgcn_s_wait_asynccnt)
#    define ASYNC_WAIT(n) __builtin_amdgcn_s_wait_asynccnt(n)
#  else
#    define ASYNC_WAIT(n) asm volatile("s_wait_asynccnt %0" ::"i"(n) : "memory")
#  endif
#else
#  define ASYNC_WAIT(n)
#endif

__device__ __forceinline__ void copy16_g2l(const unsigned short* g, unsigned short* l) {
#if HAVE_ASYNC_LDS
    __builtin_amdgcn_global_load_async_to_lds_b128(
        (__attribute__((address_space(1))) vi4_*)(void*)g,
        (__attribute__((address_space(3))) vi4_*)(void*)l, 0, 0);
#else
    *(int4*)(void*)l = *(const int4*)(const void*)g;
#endif
}

__device__ __forceinline__ unsigned short f32_to_bf16(float f) {
    union { float f; unsigned int u; } c; c.f = f;
    unsigned int u = c.u;
    u += 0x7fffu + ((u >> 16) & 1u);   // round-to-nearest-even
    return (unsigned short)(u >> 16);
}

// ---------------------------------------------------------------------------
// fp32 -> bf16 conversion
// ---------------------------------------------------------------------------
__global__ __launch_bounds__(256)
void cvt_f32_bf16_kernel(const float* __restrict__ in,
                         unsigned short* __restrict__ out, long n) {
    long i = (long)blockIdx.x * blockDim.x + threadIdx.x;
    long stride = (long)gridDim.x * blockDim.x;
    for (; i < n; i += stride) out[i] = f32_to_bf16(in[i]);
}

// ---------------------------------------------------------------------------
// Batched bf16 WMMA GEMM, LDS-tiled + double buffered:
//   C[m,n] = scale * sum_k A[m,k]*Bt[n,k] + bias[n]
// Block tile 128x64, k-step 32; 8 waves, each wave a 32x32 register tile
// (2x2 v_wmma_f32_16x16x32_bf16). A/B tiles staged global->LDS with the
// gfx1250 async-to-LDS path (ASYNCcnt) when available.
// Batch z = (zb, zh) = (z/Hdiv, z%Hdiv) with independent strides so per-(b,h)
// head views of [B,S,H*DK] tensors need no data movement.
// A/B fragments use contiguous k = 16*half + s (same in both); f32 C per CDNA5 ISA 7.12.2.
// ---------------------------------------------------------------------------
__global__ __launch_bounds__(256)
void wmma_gemm_bf16(const unsigned short* __restrict__ A, int lda, long sAb, long sAh,
                    const unsigned short* __restrict__ Bt, int ldb, long sBb, long sBh,
                    float* __restrict__ Cf, unsigned short* __restrict__ Cb,
                    int ldc, long sCb, long sCh,
                    int Hdiv, const float* __restrict__ bias, float scale, int K) {
    __shared__ unsigned short lsA[2][MT_ * KT_];   // 2 x 8 KB
    __shared__ unsigned short lsB[2][NT_ * KT_];   // 2 x 4 KB

    const int t    = threadIdx.x;
    const int lane = t & 31;
    const int wave = t >> 5;
    const int z    = blockIdx.z;
    const int zb   = z / Hdiv;
    const int zh   = z % Hdiv;

    const unsigned short* Ab = A  + zb * sAb + zh * sAh;
    const unsigned short* Bb = Bt + zb * sBb + zh * sBh;
    const long cOff          = zb * sCb + zh * sCh;

    const int m0 = blockIdx.y * MT_;
    const int n0 = blockIdx.x * NT_;

    // Staging decomposition: 16-byte chunks; A has 512 chunks (2/thread),
    // B has 256 chunks (1/thread). 3 copy instructions per wave per stage.
    const int ar0 = t >> 2;               // rows 0..63
    const int ar1 = (t + 256) >> 2;       // rows 64..127
    const int akp = (t & 3) * 8;          // k sub-offset (bf16 elements)

    auto stage = [&](int buf, int k0) {
        copy16_g2l(Ab + (long)(m0 + ar0) * lda + k0 + akp, &lsA[buf][ar0 * KT_ + akp]);
        copy16_g2l(Ab + (long)(m0 + ar1) * lda + k0 + akp, &lsA[buf][ar1 * KT_ + akp]);
        copy16_g2l(Bb + (long)(n0 + ar0) * ldb + k0 + akp, &lsB[buf][ar0 * KT_ + akp]);
    };

    // Wave -> 32x32 sub-tile of the 128x64 block tile
    const int half = lane >> 4;           // 0: lanes 0-15, 1: lanes 16-31
    const int l16  = lane & 15;
    const int mrel = (wave >> 1) * 32;
    const int nrel = (wave & 1) * 32;

    v8f acc[2][2] = {};

    stage(0, 0);
    const int nst = K / KT_;
    for (int it = 0; it < nst; ++it) {
        const int cur = it & 1;
        if (it + 1 < nst) {
            stage(cur ^ 1, (it + 1) * KT_);
            ASYNC_WAIT(3);                // retire current stage (in-order)
        } else {
            ASYNC_WAIT(0);
        }
        __syncthreads();

        v16bf af[2], bfv[2];
#pragma unroll
        for (int i = 0; i < 2; ++i) {
            const int row = mrel + i * 16 + l16;
            const __bf16* pa = (const __bf16*)&lsA[cur][row * KT_ + half * 16];
            v8bf a0 = *(const v8bf*)(pa);
            v8bf a1 = *(const v8bf*)(pa + 8);
            af[i] = __builtin_shufflevector(a0, a1,
                        0, 1, 2, 3, 4, 5, 6, 7, 8, 9, 10, 11, 12, 13, 14, 15);
        }
#pragma unroll
        for (int j = 0; j < 2; ++j) {
            const int row = nrel + j * 16 + l16;
            bfv[j] = *(const v16bf*)&lsB[cur][row * KT_ + half * 16];
        }
#pragma unroll
        for (int i = 0; i < 2; ++i)
#pragma unroll
            for (int j = 0; j < 2; ++j)
                acc[i][j] = __builtin_amdgcn_wmma_f32_16x16x32_bf16(
                                false, af[i], false, bfv[j], (short)0, acc[i][j],
                                false, false);
        __syncthreads();
    }

#pragma unroll
    for (int i = 0; i < 2; ++i) {
#pragma unroll
        for (int j = 0; j < 2; ++j) {
            const int col   = n0 + nrel + j * 16 + l16;
            const float bval = bias ? bias[col] : 0.0f;
#pragma unroll
            for (int g = 0; g < 8; ++g) {
                const int  row = m0 + mrel + i * 16 + g + half * 8;
                const long idx = cOff + (long)row * ldc + col;
                const float v  = acc[i][j][g] * scale + bval;
                if (Cb) Cb[idx] = f32_to_bf16(v);
                else    Cf[idx] = v;
            }
        }
    }
}

// ---------------------------------------------------------------------------
// qrel[b,h,l,v] = sum_d q_f32[b,l,h*DK+d] * rel_k_table[v,d]
// ---------------------------------------------------------------------------
__global__ __launch_bounds__(256)
void qrel_kernel(const float* __restrict__ q_f32, const float* __restrict__ relk,
                 float* __restrict__ qrel, long n) {
    long i = (long)blockIdx.x * blockDim.x + threadIdx.x;
    long stride = (long)gridDim.x * blockDim.x;
    for (; i < n; i += stride) {
        int v  = (int)(i % VOCAB_);
        long t = i / VOCAB_;
        int l  = (int)(t % S_);
        long bh = t / S_;           // b*H + h
        int h  = (int)(bh % H_);
        long b = bh / H_;
        const float* qp = q_f32 + (b * S_ + l) * (long)D_ + h * DK_;
        const float* rp = relk + (long)v * DK_;
        float s = 0.f;
        for (int d = 0; d < DK_; ++d) s += qp[d] * rp[d];
        qrel[i] = s;
    }
}

// ---------------------------------------------------------------------------
// vt_bf[b,h,dk,r] = v_bf[b,r,h*DK+dk]
// ---------------------------------------------------------------------------
__global__ __launch_bounds__(256)
void transpose_v_kernel(const unsigned short* __restrict__ v_bf,
                        unsigned short* __restrict__ vt_bf, long n) {
    long i = (long)blockIdx.x * blockDim.x + threadIdx.x;
    long stride = (long)gridDim.x * blockDim.x;
    for (; i < n; i += stride) {
        int r  = (int)(i % S_);
        long t = i / S_;
        int dk = (int)(t % DK_);
        long bh = t / DK_;
        int h  = (int)(bh % H_);
        long b = bh / H_;
        vt_bf[i] = v_bf[(b * S_ + r) * (long)D_ + h * DK_ + dk];
    }
}

// ---------------------------------------------------------------------------
// Row softmax over scores (in place, fp32) with relative bias added on the fly;
// also emits bf16 weights for the W@V WMMA GEMM. One block per (b,h,l) row.
// ---------------------------------------------------------------------------
__global__ __launch_bounds__(256)
void softmax_rel_kernel(float* __restrict__ weights, const float* __restrict__ qrel,
                        unsigned short* __restrict__ w_bf) {
    const int row = blockIdx.x;           // (b*H+h)*S + l
    const int l   = row % S_;
    const int t   = threadIdx.x;
    float* wrow   = weights + (long)row * S_;
    const float* qr = qrel + (long)row * VOCAB_;

    float sv[4];
    float lmax = -3.4e38f;
#pragma unroll
    for (int i = 0; i < 4; ++i) {
        int r = t + i * 256;
        int rel = r - l;
        rel = rel < -MAXREL_ ? -MAXREL_ : (rel > MAXREL_ ? MAXREL_ : rel);
        float s = wrow[r] + qr[rel + MAXREL_];
        sv[i] = s;
        lmax = fmaxf(lmax, s);
    }
    __shared__ float red[256];
    red[t] = lmax; __syncthreads();
    for (int s = 128; s > 0; s >>= 1) {
        if (t < s) red[t] = fmaxf(red[t], red[t + s]);
        __syncthreads();
    }
    const float M = red[0];
    __syncthreads();
    float lsum = 0.f;
#pragma unroll
    for (int i = 0; i < 4; ++i) { sv[i] = __expf(sv[i] - M); lsum += sv[i]; }
    red[t] = lsum; __syncthreads();
    for (int s = 128; s > 0; s >>= 1) {
        if (t < s) red[t] += red[t + s];
        __syncthreads();
    }
    const float inv = 1.0f / red[0];
#pragma unroll
    for (int i = 0; i < 4; ++i) {
        int r = t + i * 256;
        float w = sv[i] * inv;
        wrow[r] = w;
        w_bf[(long)row * S_ + r] = f32_to_bf16(w);
    }
}

// ---------------------------------------------------------------------------
// wsum[b,h,l,v] = sum over r with clip(r-l)==v-MAXREL of weights[b,h,l,r]
// ---------------------------------------------------------------------------
__global__ __launch_bounds__(256)
void wsum_kernel(const float* __restrict__ weights, float* __restrict__ wsum, long n) {
    long i = (long)blockIdx.x * blockDim.x + threadIdx.x;
    long stride = (long)gridDim.x * blockDim.x;
    for (; i < n; i += stride) {
        int v  = (int)(i % VOCAB_);
        long t = i / VOCAB_;
        int l  = (int)(t % S_);
        long bh = t / S_;
        const float* wrow = weights + (bh * S_ + l) * (long)S_;
        float s = 0.f;
        if (v == 0) {
            for (int r = 0; r <= l - MAXREL_; ++r) s += wrow[r];
        } else if (v == VOCAB_ - 1) {
            for (int r = l + MAXREL_; r < S_; ++r) s += wrow[r];
        } else {
            int r = l + v - MAXREL_;
            s = (r >= 0 && r < S_) ? wrow[r] : 0.f;
        }
        wsum[i] = s;
    }
}

// ---------------------------------------------------------------------------
// context_f32[b,l,h*DK+dk] += sum_v wsum[b,h,l,v] * rel_v_table[v,dk]
// ---------------------------------------------------------------------------
__global__ __launch_bounds__(256)
void relv_add_kernel(const float* __restrict__ wsum, const float* __restrict__ relv,
                     float* __restrict__ ctx, long n) {
    long i = (long)blockIdx.x * blockDim.x + threadIdx.x;
    long stride = (long)gridDim.x * blockDim.x;
    for (; i < n; i += stride) {
        int dk = (int)(i % DK_);
        long t = i / DK_;
        int l  = (int)(t % S_);
        long bh = t / S_;
        int h  = (int)(bh % H_);
        long b = bh / H_;
        const float* wp = wsum + (bh * S_ + l) * (long)VOCAB_;
        float s = 0.f;
#pragma unroll
        for (int v = 0; v < VOCAB_; ++v) s += wp[v] * relv[v * DK_ + dk];
        ctx[(b * S_ + l) * (long)D_ + h * DK_ + dk] += s;
    }
}

// ---------------------------------------------------------------------------
// Host driver
// ---------------------------------------------------------------------------
extern "C" void kernel_launch(void* const* d_in, const int* in_sizes, int n_in,
                              void* d_out, int out_size, void* d_ws, size_t ws_size,
                              hipStream_t stream) {
    (void)in_sizes; (void)n_in; (void)out_size; (void)ws_size;

    const float* x    = (const float*)d_in[0];
    const float* wq   = (const float*)d_in[1];
    const float* bq   = (const float*)d_in[2];
    const float* wk   = (const float*)d_in[3];
    const float* bk   = (const float*)d_in[4];
    const float* wv   = (const float*)d_in[5];
    const float* bv   = (const float*)d_in[6];
    const float* wo   = (const float*)d_in[7];
    const float* bo   = (const float*)d_in[8];
    const float* relk = (const float*)d_in[9];
    const float* relv = (const float*)d_in[10];

    float* out     = (float*)d_out;                     // [B,S,D]
    float* weights = out + (long)B_ * S_ * D_;          // [B,H,S,S] (scores in place)

    const long nXD  = (long)B_ * S_ * D_;               // 2,097,152
    const long nDD  = (long)D_ * D_;
    const long nREL = (long)B_ * H_ * S_ * VOCAB_;
    const long nW   = (long)B_ * H_ * S_ * S_;

    char* ws = (char*)d_ws;
    size_t off = 0;
    auto take = [&](size_t bytes) -> void* {
        void* p = ws + off;
        off += (bytes + 255) & ~(size_t)255;
        return p;
    };
    unsigned short* x_bf  = (unsigned short*)take(nXD * 2);
    unsigned short* wq_bf = (unsigned short*)take(nDD * 2);
    unsigned short* wk_bf = (unsigned short*)take(nDD * 2);
    unsigned short* wv_bf = (unsigned short*)take(nDD * 2);
    unsigned short* wo_bf = (unsigned short*)take(nDD * 2);
    float*          q_f32 = (float*)take(nXD * 4);
    unsigned short* q_bf  = (unsigned short*)take(nXD * 2);
    unsigned short* k_bf  = (unsigned short*)take(nXD * 2);
    unsigned short* v_bf  = (unsigned short*)take(nXD * 2);
    unsigned short* vt_bf = (unsigned short*)take(nXD * 2);
    float*          qrel  = (float*)take(nREL * 4);
    float*          wsum  = (float*)take(nREL * 4);
    unsigned short* w_bf  = (unsigned short*)take(nW * 2);
    float*          c_f32 = (float*)take(nXD * 4);
    unsigned short* c_bf  = (unsigned short*)take(nXD * 2);

    auto blocks = [](long n) { return dim3((unsigned)((n + 255) / 256)); };

    // 1) fp32 -> bf16 for activations and weights
    cvt_f32_bf16_kernel<<<blocks(nXD), 256, 0, stream>>>(x,  x_bf,  nXD);
    cvt_f32_bf16_kernel<<<blocks(nDD), 256, 0, stream>>>(wq, wq_bf, nDD);
    cvt_f32_bf16_kernel<<<blocks(nDD), 256, 0, stream>>>(wk, wk_bf, nDD);
    cvt_f32_bf16_kernel<<<blocks(nDD), 256, 0, stream>>>(wv, wv_bf, nDD);
    cvt_f32_bf16_kernel<<<blocks(nDD), 256, 0, stream>>>(wo, wo_bf, nDD);

    // 2) Projections: [4096,512] @ [512,512]^T  (torch convention: Bt = W rows)
    {
        dim3 g(D_ / NT_, (B_ * S_) / MT_, 1), b(256);
        wmma_gemm_bf16<<<g, b, 0, stream>>>(x_bf, D_, 0, 0, wq_bf, D_, 0, 0,
                                            q_f32, nullptr, D_, 0, 0,
                                            1, bq, 1.0f, D_);
        wmma_gemm_bf16<<<g, b, 0, stream>>>(x_bf, D_, 0, 0, wk_bf, D_, 0, 0,
                                            nullptr, k_bf, D_, 0, 0,
                                            1, bk, 1.0f, D_);
        wmma_gemm_bf16<<<g, b, 0, stream>>>(x_bf, D_, 0, 0, wv_bf, D_, 0, 0,
                                            nullptr, v_bf, D_, 0, 0,
                                            1, bv, 1.0f, D_);
    }
    cvt_f32_bf16_kernel<<<blocks(nXD), 256, 0, stream>>>(q_f32, q_bf, nXD);
    transpose_v_kernel<<<blocks(nXD), 256, 0, stream>>>(v_bf, vt_bf, nXD);
    qrel_kernel<<<blocks(nREL), 256, 0, stream>>>(q_f32, relk, qrel, nREL);

    // 3) scores = (Q @ K^T) * 1/sqrt(DK), batched over (b,h), into weights region
    {
        dim3 g(S_ / NT_, S_ / MT_, B_ * H_), b(256);
        wmma_gemm_bf16<<<g, b, 0, stream>>>(
            q_bf, D_, (long)S_ * D_, DK_,
            k_bf, D_, (long)S_ * D_, DK_,
            weights, nullptr, S_, (long)H_ * S_ * S_, (long)S_ * S_,
            H_, nullptr, 0.125f, DK_);
    }

    // 4) softmax (+rel_k bias) in place; bf16 copy of weights
    softmax_rel_kernel<<<dim3(B_ * H_ * S_), 256, 0, stream>>>(weights, qrel, w_bf);

    // 5) context = W @ V, batched over (b,h), written as [B,S,H*DK] fp32
    {
        dim3 g(DK_ / NT_, S_ / MT_, B_ * H_), b(256);
        wmma_gemm_bf16<<<g, b, 0, stream>>>(
            w_bf, S_, (long)H_ * S_ * S_, (long)S_ * S_,
            vt_bf, S_, (long)H_ * DK_ * S_, (long)DK_ * S_,
            c_f32, nullptr, D_, (long)S_ * D_, DK_,
            H_, nullptr, 1.0f, S_);
    }

    // 6) context += (bucketed weights) @ rel_v_table
    wsum_kernel<<<blocks(nREL), 256, 0, stream>>>(weights, wsum, nREL);
    relv_add_kernel<<<blocks(nXD), 256, 0, stream>>>(wsum, relv, c_f32, nXD);

    // 7) output = context @ Wo^T + bo
    cvt_f32_bf16_kernel<<<blocks(nXD), 256, 0, stream>>>(c_f32, c_bf, nXD);
    {
        dim3 g(D_ / NT_, (B_ * S_) / MT_, 1), b(256);
        wmma_gemm_bf16<<<g, b, 0, stream>>>(c_bf, D_, 0, 0, wo_bf, D_, 0, 0,
                                            out, nullptr, D_, 0, 0,
                                            1, bo, 1.0f, D_);
    }
}
